// SwitchLinear_43963285242755
// MI455X (gfx1250) — matrix operand order repaired and checked by reading, not compiled, not run on hardware
//
#include <hip/hip_runtime.h>

// Problem constants (match reference)
#define BB 2
#define LL 2048
#define DD 2048
#define OO 2048
#define EE 8
#define TOPK 2
#define NTOK (BB * LL)        // 4096 tokens
#define NPAIR (NTOK * TOPK)   // 8192 (token, slot) pairs

// Tiling
#define MT 128                // token-pairs per block
#define NT 64                 // output columns per block
#define KT 32                 // K per step (bf16 WMMA K)
#define LSTR 20               // LDS row stride in dwords (16 data + 4 pad, 16B aligned)

typedef __attribute__((ext_vector_type(16))) __bf16  v16bf;
typedef __attribute__((ext_vector_type(8)))  float   v8f;
typedef __attribute__((ext_vector_type(4)))  unsigned int v4u;
typedef __attribute__((ext_vector_type(4)))  float   v4f;

union Frag {
    v4u   q[2];
    v16bf v;
};

// Split two fp32 values into packed bf16 hi (truncated) and bf16 lo (rounded residual).
// Packed little-endian: element 2t -> low 16 bits of dword t.
__device__ __forceinline__ void split2(float a, float b,
                                       unsigned int& hi, unsigned int& lo) {
    unsigned int ua = __builtin_bit_cast(unsigned int, a);
    unsigned int ub = __builtin_bit_cast(unsigned int, b);
    unsigned int ha = ua >> 16;
    unsigned int hb = ub >> 16;
    float ra = a - __builtin_bit_cast(float, ha << 16);
    float rb = b - __builtin_bit_cast(float, hb << 16);
    unsigned int ura = __builtin_bit_cast(unsigned int, ra);
    unsigned int urb = __builtin_bit_cast(unsigned int, rb);
    unsigned int la = ((ura + 0x7FFFu + ((ura >> 16) & 1u)) >> 16) & 0xFFFFu;
    unsigned int lb = ((urb + 0x7FFFu + ((urb >> 16) & 1u)) >> 16) & 0xFFFFu;
    hi = ha | (hb << 16);
    lo = la | (lb << 16);
}

__global__ __launch_bounds__(32) void zero_counts_kernel(unsigned int* cnt) {
    if (threadIdx.x < EE) cnt[threadIdx.x] = 0u;
}

// Bucket (token, slot) pairs by expert. Output identity is keyed by pair id,
// so atomic ordering does not affect the final d_out contents (deterministic).
__global__ __launch_bounds__(256) void bin_kernel(const int* __restrict__ idx,
                                                  unsigned int* __restrict__ cnt,
                                                  unsigned int* __restrict__ list) {
    int p = blockIdx.x * 256 + threadIdx.x;
    if (p < NPAIR) {
        int e = idx[p];
        unsigned int pos = atomicAdd(&cnt[e], 1u);
        list[(size_t)e * NPAIR + pos] = (unsigned int)p;
    }
}

// Grouped GEMM: per expert e, gathered tokens (MT per block) x weight[e] tile.
// bf16x3 fp32 emulation: acc += Ah*Bl + Al*Bh + Ah*Bh via v_wmma_f32_16x16x32_bf16.
__global__ __launch_bounds__(256) void moe_gemm_kernel(
    const float* __restrict__ x,     // (NTOK, D)
    const float* __restrict__ w,     // (E, O, D)
    const float* __restrict__ bias,  // (E, O)
    const unsigned int* __restrict__ cnt,
    const unsigned int* __restrict__ list,
    float* __restrict__ out) {       // (NPAIR, O)

    const int e = blockIdx.z;
    const unsigned int c = cnt[e];
    const int mtile = blockIdx.y;
    if ((unsigned int)(mtile * MT) >= c) return;   // uniform early exit
    const int obase = blockIdx.x * NT;

    __shared__ unsigned int sA_hi[MT * LSTR];
    __shared__ unsigned int sA_lo[MT * LSTR];
    __shared__ unsigned int sB_hi[NT * LSTR];
    __shared__ unsigned int sB_lo[NT * LSTR];
    __shared__ unsigned int sPair[MT];

    const int tid = threadIdx.x;
    if (tid < MT) {
        unsigned int gr = (unsigned int)(mtile * MT + tid);
        sPair[tid] = (gr < c) ? list[(size_t)e * NPAIR + gr] : 0xFFFFFFFFu;
    }
    __syncthreads();

    // ---- Loop-invariant loader setup (hoisted: no per-step LDS reads/branches) ----
    // A loader: 8 threads per row, 4 floats each; thread owns 4 rows.
    const int rA  = tid >> 3;              // base row (0..31), +32*it
    const int d2A = (tid & 7) * 2;         // packed-dword offset in LDS row
    const float* aptr[4];
#pragma unroll
    for (int it = 0; it < 4; ++it) {
        unsigned int p = sPair[rA + it * 32];
        // Invalid rows clamp to token 0: their garbage only reaches C-rows that
        // are never stored (each A-row feeds only its own output row).
        unsigned int tok = (p != 0xFFFFFFFFu) ? (p >> 1) : 0u;
        aptr[it] = x + (size_t)tok * DD + (size_t)((tid & 7) * 4);
    }
    // B loader: 4 threads per row, 8 floats each.
    const int rowB = tid >> 2;             // 0..63
    const int d2B  = (tid & 3) * 4;
    const float* bptr = w + ((size_t)e * OO + (size_t)(obase + rowB)) * DD
                          + (size_t)((tid & 3) * 8);

    const int lane = tid & 31;
    const int wav  = tid >> 5;     // 0..7
    const int wm   = wav & 3;      // wave M offset: 32*wm
    const int wn   = wav >> 2;     // wave N offset: 32*wn
    const int g    = lane >> 4;    // lane group (0/1)
    const int l16  = lane & 15;

    v8f acc[2][2];
#pragma unroll
    for (int i = 0; i < 2; ++i)
#pragma unroll
        for (int j = 0; j < 2; ++j)
#pragma unroll
            for (int t = 0; t < 8; ++t) acc[i][j][t] = 0.0f;

    // ---- One-stage register pipeline: global loads for step k+1 issue while
    //      WMMAs for step k run; LOADcnt wait lands at the next store_tiles. ----
    v4f aReg[4];
    v4f bReg0, bReg1;

    auto load_tiles = [&](int k0) {
#pragma unroll
        for (int it = 0; it < 4; ++it)
            aReg[it] = *(const v4f*)(aptr[it] + k0);
        bReg0 = *(const v4f*)(bptr + k0);
        bReg1 = *(const v4f*)(bptr + k0 + 4);
        if (k0 + KT < DD) __builtin_prefetch(bptr + k0 + KT, 0, 0);  // global_prefetch_b8
    };

    auto store_tiles = [&]() {
#pragma unroll
        for (int it = 0; it < 4; ++it) {
            unsigned int h0, l0, h1, l1;
            split2(aReg[it][0], aReg[it][1], h0, l0);
            split2(aReg[it][2], aReg[it][3], h1, l1);
            int base = (rA + it * 32) * LSTR + d2A;
            sA_hi[base]     = h0;
            sA_hi[base + 1] = h1;
            sA_lo[base]     = l0;
            sA_lo[base + 1] = l1;
        }
        unsigned int h0, l0, h1, l1, h2, l2, h3, l3;
        split2(bReg0[0], bReg0[1], h0, l0);
        split2(bReg0[2], bReg0[3], h1, l1);
        split2(bReg1[0], bReg1[1], h2, l2);
        split2(bReg1[2], bReg1[3], h3, l3);
        int base = rowB * LSTR + d2B;
        sB_hi[base]     = h0; sB_hi[base + 1] = h1;
        sB_hi[base + 2] = h2; sB_hi[base + 3] = h3;
        sB_lo[base]     = l0; sB_lo[base + 1] = l1;
        sB_lo[base + 2] = l2; sB_lo[base + 3] = l3;
    };

    load_tiles(0);

    for (int k0 = 0; k0 < DD; k0 += KT) {
        store_tiles();          // consumes regs loaded last iteration
        __syncthreads();

        if (k0 + KT < DD) load_tiles(k0 + KT);   // overlap with WMMAs below

        // ---- Fragment fetch from LDS (documented CDNA5 16-bit layouts) ----
        // A (M x K): lane m = l16; slots s<8 -> K = s + 8g, s>=8 -> K = s + 8 + 8g
        Frag ah[2], al[2], bh[2], bl[2];
#pragma unroll
        for (int i = 0; i < 2; ++i) {
            int m = wm * 32 + i * 16 + l16;
            const unsigned int* rh = &sA_hi[m * LSTR];
            const unsigned int* rl = &sA_lo[m * LSTR];
            ah[i].q[0] = *(const v4u*)(rh + 4 * g);
            ah[i].q[1] = *(const v4u*)(rh + 8 + 4 * g);
            al[i].q[0] = *(const v4u*)(rl + 4 * g);
            al[i].q[1] = *(const v4u*)(rl + 8 + 4 * g);
        }
        // B (K x N): lane n = l16; slots s<8 -> K = s + 8g, s>=8 -> K = s + 8 + 8g
#pragma unroll
        for (int j = 0; j < 2; ++j) {
            int n = wn * 32 + j * 16 + l16;
            const unsigned int* rh = &sB_hi[n * LSTR];
            const unsigned int* rl = &sB_lo[n * LSTR];
            bh[j].q[0] = *(const v4u*)(rh + 4 * g);
            bh[j].q[1] = *(const v4u*)(rh + 8 + 4 * g);
            bl[j].q[0] = *(const v4u*)(rl + 4 * g);
            bl[j].q[1] = *(const v4u*)(rl + 8 + 4 * g);
        }

        // ---- bf16x3 WMMA: Ah*Bl + Al*Bh + Ah*Bh ----
#pragma unroll
        for (int i = 0; i < 2; ++i)
#pragma unroll
            for (int j = 0; j < 2; ++j) {
                acc[i][j] = __builtin_amdgcn_wmma_f32_16x16x32_bf16(
                    false, ah[i].v, false, bl[j].v, (short)0, acc[i][j], false, false);
                acc[i][j] = __builtin_amdgcn_wmma_f32_16x16x32_bf16(
                    false, al[i].v, false, bh[j].v, (short)0, acc[i][j], false, false);
                acc[i][j] = __builtin_amdgcn_wmma_f32_16x16x32_bf16(
                    false, ah[i].v, false, bh[j].v, (short)0, acc[i][j], false, false);
            }

        __syncthreads();
    }

    // ---- Epilogue: C/D layout lane n = l16, VGPR r -> M = r + 8g; add bias, scatter ----
#pragma unroll
    for (int j = 0; j < 2; ++j) {
        int n = obase + wn * 32 + j * 16 + l16;
        float bv = bias[e * OO + n];
#pragma unroll
        for (int i = 0; i < 2; ++i) {
#pragma unroll
            for (int r = 0; r < 8; ++r) {
                int mloc = wm * 32 + i * 16 + r + 8 * g;
                unsigned int p = sPair[mloc];
                if (p != 0xFFFFFFFFu) {
                    out[(size_t)p * OO + (size_t)n] = acc[i][j][r] + bv;
                }
            }
        }
    }
}

extern "C" void kernel_launch(void* const* d_in, const int* in_sizes, int n_in,
                              void* d_out, int out_size, void* d_ws, size_t ws_size,
                              hipStream_t stream) {
    const float* x    = (const float*)d_in[0];  // (B, L, D)
    const float* w    = (const float*)d_in[1];  // (E, O, D)
    const float* bias = (const float*)d_in[2];  // (E, O)
    const int*   idx  = (const int*)d_in[3];    // (B, L, K)
    float* out = (float*)d_out;                 // (B, L, K, O)

    unsigned int* cnt  = (unsigned int*)d_ws;   // 8 counters
    unsigned int* list = cnt + 8;               // 8 * 8192 pair ids (~256 KB)

    zero_counts_kernel<<<1, 32, 0, stream>>>(cnt);
    bin_kernel<<<(NPAIR + 255) / 256, 256, 0, stream>>>(idx, cnt, list);

    dim3 grid(OO / NT, (NPAIR + MT - 1) / MT, EE);
    moe_gemm_kernel<<<grid, 256, 0, stream>>>(x, w, bias, cnt, list, out);
}
